// DGN_critic_56143812493412
// MI455X (gfx1250) — matrix operand order repaired and checked by reading, not compiled, not run on hardware
//
#include <hip/hip_runtime.h>

typedef __attribute__((ext_vector_type(16))) _Float16 v16h;
typedef __attribute__((ext_vector_type(8)))  _Float16 v8h;
typedef __attribute__((ext_vector_type(4)))  _Float16 v4h;
typedef __attribute__((ext_vector_type(8)))  float    v8f;

// ---------------------------------------------------------------------------
// A-fragment loader for V_WMMA_F32_16X16X32_F16 (ISA 7.12.2, 16-bit A 16x32):
// lane = (half)*16 + m%16 ; per lane the 16 halves are contiguous
// at K offsets half*16 .. half*16+15 (this code's k order, shared with B; not the ISA table order).
// ---------------------------------------------------------------------------
static __device__ __forceinline__ v16h make_a_frag(const _Float16* arow, int kt) {
  v8h lo = *(const v8h*)(arow + kt * 32);
  v8h hi = *(const v8h*)(arow + kt * 32 + 8);
  return __builtin_shufflevector(lo, hi, 0,1,2,3,4,5,6,7,8,9,10,11,12,13,14,15);
}

// ---------------------------------------------------------------------------
// One wave computes its 16-row tile x all 16 col tiles of OUT[128][256]:
//   OUT = act_fn( A[128][KT*32] @ W^T + bias (+ act*w1a) )
// A in LDS (f16), W row-major [256][KT*32] (global f16 or LDS f16), OUT LDS f16.
// A-fragments hoisted to registers once (KT * 8 VGPRs).
// TRANS=true stores OUT column-major; the 8 row elements per lane are then
// contiguous -> packed single ds_store_b128 per tile.
// ---------------------------------------------------------------------------
template<int KT, bool RELU, bool TRANS, bool AFF, bool BIAS>
static __device__ __forceinline__
void gemm_tile(const _Float16* __restrict__ A, int lda,
               const _Float16* __restrict__ W, int ldb,
               const float* __restrict__ bias,
               _Float16* __restrict__ O, int ldo,
               int wave, int lane,
               const float* __restrict__ act, const float* __restrict__ w1a)
{
  const int ln = lane & 15;
  const int lh = lane >> 4;
  const _Float16* arow = A + (wave * 16 + ln) * lda + lh * 16;
  v16h af[KT];
#pragma unroll
  for (int kt = 0; kt < KT; ++kt) af[kt] = make_a_frag(arow, kt);

  for (int ct = 0; ct < 16; ++ct) {
    const int col = ct * 16 + ln;
    const _Float16* bcol = W + (size_t)col * ldb + lh * 16;
    v8f acc = {0.f, 0.f, 0.f, 0.f, 0.f, 0.f, 0.f, 0.f};
#pragma unroll
    for (int kt = 0; kt < KT; ++kt) {
      v16h bf = *(const v16h*)(bcol + kt * 32);
      acc = __builtin_amdgcn_wmma_f32_16x16x32_f16(false, af[kt], false, bf,
                                                   (short)0, acc, false, false);
    }
    const float bs = BIAS ? bias[col] : 0.f;
    const float wa = AFF ? w1a[(size_t)col * 257] : 0.f;
    if (TRANS) {
      v8h pk;
#pragma unroll
      for (int j = 0; j < 8; ++j) {
        float r = acc[j] + bs;
        if (RELU) r = fmaxf(r, 0.f);
        pk[j] = (_Float16)r;
      }
      *(v8h*)(O + (size_t)col * ldo + wave * 16 + lh * 8) = pk;   // ds_store_b128
    } else {
#pragma unroll
      for (int j = 0; j < 8; ++j) {
        const int row = wave * 16 + lh * 8 + j;
        float r = acc[j] + bs;
        if (AFF)  r += act[row] * wa;
        if (RELU) r = fmaxf(r, 0.f);
        O[(size_t)row * ldo + col] = (_Float16)r;
      }
    }
  }
}

// ---------------------------------------------------------------------------
// Fused DGN critic: one workgroup per batch element, 8 wave32 per workgroup,
// 256 KB dynamic LDS carved into 4 x 64KB recycled slots.
// ---------------------------------------------------------------------------
__global__ __launch_bounds__(256, 1)
void dgn_fused_kernel(
    const float* __restrict__ x, const float* __restrict__ mask,
    const float* __restrict__ action,
    const float* __restrict__ be, const float* __restrict__ bv,
    const float* __restrict__ bk, const float* __restrict__ bq,
    const float* __restrict__ W1full, const float* __restrict__ b1,
    const float* __restrict__ b2, const float* __restrict__ b3,
    const float* __restrict__ W4, const float* __restrict__ b4,
    const _Float16* __restrict__ We16, const _Float16* __restrict__ Wv16,
    const _Float16* __restrict__ Wk16, const _Float16* __restrict__ Wq16,
    const _Float16* __restrict__ W1h16, const _Float16* __restrict__ W2_16,
    const _Float16* __restrict__ W3_16,
    float* __restrict__ out_value, float* __restrict__ out_aw)
{
  extern __shared__ char smem[];
  _Float16* slot0 = (_Float16*)(smem);                // h1 -> c1
  _Float16* slot1 = (_Float16*)(smem + 64 * 1024);    // q  -> att -> c2
  _Float16* slot2 = (_Float16*)(smem + 128 * 1024);   // k  -> h2  -> c3
  _Float16* slot3 = (_Float16*)(smem + 192 * 1024);   // x  -> v^T

  const int b    = blockIdx.x;
  const int tid  = threadIdx.x;
  const int lane = tid & 31;
  const int wave = tid >> 5;        // wave w owns row tile w (rows 16w..16w+15)
  const int ln   = lane & 15;
  const int lh   = lane >> 4;

  // Phase 0: stage x[b] (f32 -> f16) into LDS [128][64], vectorized
  {
    const float4* xb4 = (const float4*)(x + (size_t)b * 128 * 64);
    v4h* s4 = (v4h*)slot3;
    for (int i = tid; i < 128 * 64 / 4; i += 256) {
      float4 f = xb4[i];
      v4h h;
      h[0] = (_Float16)f.x; h[1] = (_Float16)f.y;
      h[2] = (_Float16)f.z; h[3] = (_Float16)f.w;
      s4[i] = h;
    }
  }
  __syncthreads();

  // Phase 1: h1 = relu(x @ We^T + be) -> slot0 [128][256]
  gemm_tile<2, true, false, false, true>(slot3, 64, We16, 64, be, slot0, 256,
                                         wave, lane, nullptr, nullptr);
  __syncthreads();

  // Phase 2: q,k,v projections (v stored transposed [256][128] for att@v)
  gemm_tile<8, true, false, false, true>(slot0, 256, Wq16, 256, bq, slot1, 256,
                                         wave, lane, nullptr, nullptr);
  gemm_tile<8, true, false, false, true>(slot0, 256, Wk16, 256, bk, slot2, 256,
                                         wave, lane, nullptr, nullptr);
  gemm_tile<8, true, true,  false, true>(slot0, 256, Wv16, 256, bv, slot3, 128,
                                         wave, lane, nullptr, nullptr);
  __syncthreads();

  // Phase 3: scores = q @ k^T kept in registers; mask -> a_w (global);
  // register softmax; att (f16) -> slot1 [128][128]
  {
    const _Float16* qrow = slot1 + (wave * 16 + ln) * 256 + lh * 16;
    v16h qf[8];
#pragma unroll
    for (int kt = 0; kt < 8; ++kt) qf[kt] = make_a_frag(qrow, kt);
    __syncthreads();   // all q-fragments hoisted; slot1 reusable for att

    v8f sc[8];
#pragma unroll
    for (int ct = 0; ct < 8; ++ct) {
      const _Float16* kcol = slot2 + (ct * 16 + ln) * 256 + lh * 16;
      v8f acc = {0.f, 0.f, 0.f, 0.f, 0.f, 0.f, 0.f, 0.f};
#pragma unroll
      for (int kt = 0; kt < 8; ++kt) {
        v16h bf = *(const v16h*)(kcol + kt * 32);
        acc = __builtin_amdgcn_wmma_f32_16x16x32_f16(false, qf[kt], false, bf,
                                                     (short)0, acc, false, false);
      }
      sc[ct] = acc;
    }

    float aw[8][8];
    const float* mb  = mask   + (size_t)b * 128 * 128;
    float*       awb = out_aw + (size_t)b * 128 * 128;
#pragma unroll
    for (int ct = 0; ct < 8; ++ct) {
      const int col = ct * 16 + ln;
#pragma unroll
      for (int j = 0; j < 8; ++j) {
        const int row = wave * 16 + lh * 8 + j;
        float m = mb[row * 128 + col];
        float a = fminf(fmaxf(sc[ct][j] * m, 0.f), 9e13f) - 9e15f * (1.f - m);
        aw[ct][j] = a;
        awb[row * 128 + col] = a;          // streaming f32 output
      }
    }

    // softmax over 128 cols of each row: 8 in-lane tiles + 16-lane half reduce
#pragma unroll
    for (int j = 0; j < 8; ++j) {
      float mx = aw[0][j];
#pragma unroll
      for (int ct = 1; ct < 8; ++ct) mx = fmaxf(mx, aw[ct][j]);
      for (int d = 1; d < 16; d <<= 1) mx = fmaxf(mx, __shfl_xor(mx, d, 32));
      float sum = 0.f;
#pragma unroll
      for (int ct = 0; ct < 8; ++ct) {
        float e = __expf(aw[ct][j] - mx);
        aw[ct][j] = e;
        sum += e;
      }
      for (int d = 1; d < 16; d <<= 1) sum += __shfl_xor(sum, d, 32);
      const float inv = 1.f / sum;
      const int row = wave * 16 + lh * 8 + j;
#pragma unroll
      for (int ct = 0; ct < 8; ++ct)
        slot1[row * 128 + ct * 16 + ln] = (_Float16)(aw[ct][j] * inv);
    }
  }
  __syncthreads();

  // Phase 4: h2 = att @ v  (B operand = v^T in slot3, contiguous) -> slot2
  gemm_tile<4, false, false, false, false>(slot1, 128, slot3, 128, nullptr,
                                           slot2, 256, wave, lane, nullptr, nullptr);
  __syncthreads();

  // Phase 5: c1 = relu([h2, action] @ W1^T + b1) -> slot0
  gemm_tile<8, true, false, true, true>(slot2, 256, W1h16, 256, b1, slot0, 256,
                                        wave, lane, action + (size_t)b * 128,
                                        W1full + 256 /* last col, stride 257 */);
  __syncthreads();

  // Phase 6: c2 = relu(c1 @ W2^T + b2) -> slot1
  gemm_tile<8, true, false, false, true>(slot0, 256, W2_16, 256, b2, slot1, 256,
                                         wave, lane, nullptr, nullptr);
  __syncthreads();

  // Phase 7: c3 = relu(c2 @ W3^T + b3) -> slot2
  gemm_tile<8, true, false, false, true>(slot1, 256, W3_16, 256, b3, slot2, 256,
                                         wave, lane, nullptr, nullptr);
  __syncthreads();

  // Phase 8: value[n] = c3[n] . W4 + b4  (thin GEMV, vectorized VALU)
  if (tid < 128) {
    const v8h*    crow = (const v8h*)(slot2 + tid * 256);
    const float4* w4   = (const float4*)W4;
    float acc = 0.f;
#pragma unroll 4
    for (int c8 = 0; c8 < 32; ++c8) {
      v8h h = crow[c8];
      float4 wa = w4[2 * c8];
      float4 wb = w4[2 * c8 + 1];
      acc += (float)h[0] * wa.x + (float)h[1] * wa.y +
             (float)h[2] * wa.z + (float)h[3] * wa.w +
             (float)h[4] * wb.x + (float)h[5] * wb.y +
             (float)h[6] * wb.z + (float)h[7] * wb.w;
    }
    out_value[(size_t)b * 128 + tid] = acc + b4[0];
  }
}

// ---------------------------------------------------------------------------
// Weight pre-conversion f32 -> f16 into workspace (runs once per launch;
// weights then stay L2-resident and are broadcast to all 1024 blocks).
// ---------------------------------------------------------------------------
__global__ void cvt_f16_kernel(const float* __restrict__ s,
                               _Float16* __restrict__ d, int n) {
  int i = blockIdx.x * blockDim.x + threadIdx.x;
  if (i < n) d[i] = (_Float16)s[i];
}

// W1 is [256][257]; extract first 256 cols of each row as dense [256][256]
__global__ void cvt_W1h_kernel(const float* __restrict__ W1,
                               _Float16* __restrict__ d) {
  int i = blockIdx.x * blockDim.x + threadIdx.x;
  if (i < 256 * 256) d[i] = (_Float16)W1[(i >> 8) * 257 + (i & 255)];
}

extern "C" void kernel_launch(void* const* d_in, const int* in_sizes, int n_in,
                              void* d_out, int out_size, void* d_ws, size_t ws_size,
                              hipStream_t stream) {
  (void)in_sizes; (void)n_in; (void)out_size; (void)ws_size;
  const float* x      = (const float*)d_in[0];
  const float* mask   = (const float*)d_in[1];
  const float* action = (const float*)d_in[2];
  const float* We     = (const float*)d_in[3];
  const float* be     = (const float*)d_in[4];
  const float* Wv     = (const float*)d_in[5];
  const float* bv     = (const float*)d_in[6];
  const float* Wk     = (const float*)d_in[7];
  const float* bk     = (const float*)d_in[8];
  const float* Wq     = (const float*)d_in[9];
  const float* bq     = (const float*)d_in[10];
  const float* W1     = (const float*)d_in[11];
  const float* b1     = (const float*)d_in[12];
  const float* W2     = (const float*)d_in[13];
  const float* b2     = (const float*)d_in[14];
  const float* W3     = (const float*)d_in[15];
  const float* b3     = (const float*)d_in[16];
  const float* W4     = (const float*)d_in[17];
  const float* b4     = (const float*)d_in[18];

  // f16 weight arena in workspace (819,200 bytes total)
  _Float16* We16  = (_Float16*)d_ws;          // 256*64
  _Float16* Wv16  = We16  + 256 * 64;         // 256*256
  _Float16* Wk16  = Wv16  + 256 * 256;
  _Float16* Wq16  = Wk16  + 256 * 256;
  _Float16* W1h16 = Wq16  + 256 * 256;
  _Float16* W2_16 = W1h16 + 256 * 256;
  _Float16* W3_16 = W2_16 + 256 * 256;

  cvt_f16_kernel<<<(256 * 64 + 255) / 256, 256, 0, stream>>>(We, We16, 256 * 64);
  cvt_f16_kernel<<<256, 256, 0, stream>>>(Wv, Wv16, 256 * 256);
  cvt_f16_kernel<<<256, 256, 0, stream>>>(Wk, Wk16, 256 * 256);
  cvt_f16_kernel<<<256, 256, 0, stream>>>(Wq, Wq16, 256 * 256);
  cvt_W1h_kernel<<<256, 256, 0, stream>>>(W1, W1h16);
  cvt_f16_kernel<<<256, 256, 0, stream>>>(W2, W2_16, 256 * 256);
  cvt_f16_kernel<<<256, 256, 0, stream>>>(W3, W3_16, 256 * 256);

  float* out_value = (float*)d_out;                 // [1024*128]
  float* out_aw    = out_value + 1024 * 128;        // [1024*128*128]

  dgn_fused_kernel<<<1024, 256, 256 * 1024, stream>>>(
      x, mask, action, be, bv, bk, bq, W1, b1, b2, b3, W4, b4,
      We16, Wv16, Wk16, Wq16, W1h16, W2_16, W3_16,
      out_value, out_aw);
}
